// Model_33071248179898
// MI455X (gfx1250) — compile-verified, matrix operand order checked by reading, not run on hardware
//
#include <hip/hip_runtime.h>

// ---------------------------------------------------------------------------
// Per-pixel patch-MLP (25->50->75->75->50->1) with per-channel global BatchNorm
// between layers, mapped to CDNA5 wave32 WMMA (v_wmma_f32_16x16x32_f16).
//
// Design (MI455X roofline): ~45 GFLOP of GEMM vs ~2 GB of f16 intermediate
// HBM traffic (BN needs global stats -> intermediates must round-trip HBM,
// they exceed the 192MB L2). Memory bound at ~90us @ 23.3 TB/s, so:
//   * f16 WMMA with f32 accumulation (compute is then free),
//   * BN stats (sum, sumsq per channel) fused into GEMM epilogue via
//     LDS reduction + global f32 atomics,
//   * BN normalize + ReLU fused into the NEXT layer's A-fragment load,
//   * weights prepacked into WMMA B-fragment layout, zero-padded so K/N
//     padding contributes exactly zero.
// ---------------------------------------------------------------------------

typedef _Float16 f16;
typedef __attribute__((ext_vector_type(16))) _Float16 v16h;
typedef __attribute__((ext_vector_type(8)))  _Float16 v8h;
typedef __attribute__((ext_vector_type(8)))  float    v8f;

#define RTOT   (8 * 3 * 256 * 256)   // 1,572,864 rows (b,c,h,w)
#define EPSV   1e-5f
#define PB_STRIDE 10240              // halves per layer slot in packB

__device__ __forceinline__ int reflect256(int i) {
  if (i < 0) i = -i;
  if (i > 255) i = 510 - i;
  return i;
}

// ---------------------------------------------------------------------------
// Pack W[l] (fo x fi, row-major) into WMMA B fragments (K x N = 32 x 16 tiles),
// f16, zero padded. Fragment element layout (wave32, 16-bit B):
//   lane L: N = L%16 ; base = (L<16 ? 0 : 8) ; K(e) = kt*32 + base + (e<8?e:e+8)
// Also pad biases to 96 floats per layer.
// ---------------------------------------------------------------------------
__global__ void prepack_kernel(const float* w1, const float* b1,
                               const float* w2, const float* b2,
                               const float* w3, const float* b3,
                               const float* w4, const float* b4,
                               const float* w5, const float* b5,
                               f16* __restrict__ packB, float* __restrict__ biasPad) {
  const float* W[5] = {w1, w2, w3, w4, w5};
  const float* B[5] = {b1, b2, b3, b4, b5};
  const int FI[5] = {25, 50, 75, 75, 50};
  const int FO[5] = {50, 75, 75, 50, 1};
  for (int l = 0; l < 5; ++l) {
    const int KP = (FI[l] + 31) / 32 * 32, KT = KP / 32;
    const int NP = (FO[l] + 31) / 32 * 32, NT = NP / 16;
    const int total = KT * NT * 512;
    f16* pb = packB + l * PB_STRIDE;
    for (int idx = threadIdx.x; idx < total; idx += blockDim.x) {
      int e    = idx & 15;
      int lane = (idx >> 4) & 31;
      int frag = idx >> 9;
      int kt = frag / NT, nt = frag % NT;
      int n    = nt * 16 + (lane & 15);
      int base = (lane < 16) ? 0 : 8;
      int k    = kt * 32 + base + (e < 8 ? e : e + 8);
      float v  = (n < FO[l] && k < FI[l]) ? W[l][n * FI[l] + k] : 0.0f;
      pb[idx] = (f16)v;
    }
    for (int n = threadIdx.x; n < 96; n += blockDim.x)
      biasPad[l * 96 + n] = (n < FO[l]) ? B[l][n] : 0.0f;
  }
}

// ---------------------------------------------------------------------------
// Shared GEMM epilogue: add bias, accumulate per-channel sum/sumsq over valid
// columns (n < FO), store pre-norm y as f16 (width WOUT), reduce stats in LDS
// then one global atomicAdd pair per block. All 128 rows of a block share one
// image channel (65536 rows per (b,c) plane, 128 | 65536).
// C/D layout: VGPR j, lanes 0-15 -> M=j, lanes 16-31 -> M=j+8; N = lane%16.
// ---------------------------------------------------------------------------
template <int FO, int NT, int WOUT>
__device__ __forceinline__ void epilogue_store(v8f* acc, int rowTile, int lane,
                                               const float* __restrict__ biasPad,
                                               f16* __restrict__ yout,
                                               float* __restrict__ statsOut, int c) {
  float lsum = 0.f, lsq = 0.f;
  const int nlo = lane & 15;
  const int mhi = (lane >> 4) * 8;
#pragma unroll
  for (int nt = 0; nt < NT; ++nt) {
    const int n = nt * 16 + nlo;
    const float bias = biasPad[n];
#pragma unroll
    for (int j = 0; j < 8; ++j) {
      float v = acc[nt][j] + bias;
      if (n < FO) { lsum += v; lsq += v * v; }
      if (n < WOUT) yout[(size_t)(rowTile + mhi + j) * WOUT + n] = (f16)v;
    }
  }
  __shared__ float red[2];
  if (threadIdx.x < 2) red[threadIdx.x] = 0.f;
  __syncthreads();
  atomicAdd(&red[0], lsum);   // ds_add_f32
  atomicAdd(&red[1], lsq);
  __syncthreads();
  if (threadIdx.x == 0) {
    atomicAdd(&statsOut[c],     red[0]);   // global_atomic_add_f32
    atomicAdd(&statsOut[3 + c], red[1]);
  }
}

// ---------------------------------------------------------------------------
// Layer 1: A fragments gathered from the input image with reflect padding
// (x is 6 MB -> L2 resident), FI=25 (K padded to 32), FO=50 (N padded to 64).
// ---------------------------------------------------------------------------
__global__ void __launch_bounds__(256)
gemm_layer1(const float* __restrict__ x, f16* __restrict__ yout,
            const f16* __restrict__ packB, const float* __restrict__ biasPad,
            float* __restrict__ statsOut) {
  constexpr int FO = 50, NT = 4, WOUT = 64;  // KP=32 -> KT=1
  const int lane = threadIdx.x & 31;
  const int wave = threadIdx.x >> 5;
  const int rowTile = blockIdx.x * 128 + wave * 16;
  const int row = rowTile + (lane & 15);
  const int bb = row / 196608;
  const int c  = (row >> 16) % 3;
  const int h  = (row >> 8) & 255;
  const int w  = row & 255;
  const float* xp = x + (size_t)(bb * 3 + c) * 65536;
  const int base = (lane & 16) ? 8 : 0;

  v16h a;
#pragma unroll
  for (int e = 0; e < 16; ++e) {
    const int k = base + (e < 8 ? e : e + 8);
    float v = 0.0f;
    if (k < 25) {
      const int hh = reflect256(h + k / 5 - 2);
      const int ww = reflect256(w + k % 5 - 2);
      v = xp[hh * 256 + ww];
    }
    a[e] = (f16)v;
  }

  v8f acc[NT];
  const v8f zero = {};
#pragma unroll
  for (int nt = 0; nt < NT; ++nt) acc[nt] = zero;

  const f16* pb = packB + lane * 16;
#pragma unroll
  for (int nt = 0; nt < NT; ++nt) {
    v16h b = *(const v16h*)(pb + nt * 512);
    acc[nt] = __builtin_amdgcn_wmma_f32_16x16x32_f16(false, a, false, b,
                                                     (short)0, acc[nt], false, false);
  }
  epilogue_store<FO, NT, WOUT>(acc, rowTile, lane, biasPad, yout, statsOut, c);
}

// ---------------------------------------------------------------------------
// Layers 2..5: read previous pre-norm f16 activations (width WPREV == pad32(FI)
// so all 16-byte half8 loads are in-bounds and aligned), fuse BN(prev)+ReLU
// into the A-fragment build, GEMM into NT column tiles, shared epilogue.
// ---------------------------------------------------------------------------
template <int FI, int FO, int WPREV, int WOUT>
__global__ void __launch_bounds__(256)
gemm_layer(const f16* __restrict__ yprev, f16* __restrict__ yout,
           const f16* __restrict__ packB, const float* __restrict__ biasPad,
           const float* __restrict__ statsPrev, const float* __restrict__ gPrev,
           const float* __restrict__ bePrev, float* __restrict__ statsOut,
           float invCntPrev) {
  constexpr int KP = (FI + 31) / 32 * 32, KT = KP / 32;
  constexpr int NP = (FO + 31) / 32 * 32, NT = NP / 16;
  const int lane = threadIdx.x & 31;
  const int wave = threadIdx.x >> 5;
  const int rowTile = blockIdx.x * 128 + wave * 16;
  const int c = (rowTile >> 16) % 3;

  // BN of previous layer: a = relu(y*rs + sh)
  const float mean = statsPrev[c] * invCntPrev;
  const float var  = statsPrev[3 + c] * invCntPrev - mean * mean;
  const float rs   = rsqrtf(var + EPSV) * gPrev[c];
  const float sh   = bePrev[c] - mean * rs;

  v8f acc[NT];
  const v8f zero = {};
#pragma unroll
  for (int nt = 0; nt < NT; ++nt) acc[nt] = zero;

  const int row  = rowTile + (lane & 15);
  const int base = (lane & 16) ? 8 : 0;
  const f16* rp = yprev + (size_t)row * WPREV + base;

#pragma unroll
  for (int kt = 0; kt < KT; ++kt) {
    const v8h c0 = *(const v8h*)(rp + kt * 32);        // K = kt*32+base+0..7
    const v8h c1 = *(const v8h*)(rp + kt * 32 + 16);   // K = kt*32+base+16..23
    v16h a;
#pragma unroll
    for (int e = 0; e < 8; ++e) {
      a[e]     = (f16)fmaxf(0.0f, (float)c0[e] * rs + sh);
      a[e + 8] = (f16)fmaxf(0.0f, (float)c1[e] * rs + sh);
    }
    const f16* pb = packB + (size_t)(kt * NT) * 512 + lane * 16;
#pragma unroll
    for (int nt = 0; nt < NT; ++nt) {
      v16h b = *(const v16h*)(pb + nt * 512);
      acc[nt] = __builtin_amdgcn_wmma_f32_16x16x32_f16(false, a, false, b,
                                                       (short)0, acc[nt], false, false);
    }
  }
  epilogue_store<FO, NT, WOUT>(acc, rowTile, lane, biasPad, yout, statsOut, c);
}

// ---------------------------------------------------------------------------
// Final BN + ReLU of y5 (single feature) into f32 output.
// ---------------------------------------------------------------------------
__global__ void __launch_bounds__(256)
finalize_kernel(const f16* __restrict__ y5, const float* __restrict__ stats,
                const float* __restrict__ g, const float* __restrict__ be,
                float* __restrict__ out) {
  const int r = blockIdx.x * 256 + threadIdx.x;
  const int c = (r >> 16) % 3;
  const float inv  = 1.0f / 524288.0f;   // b*h*w per channel, fo=1
  const float mean = stats[c] * inv;
  const float var  = stats[3 + c] * inv - mean * mean;
  const float rs   = rsqrtf(var + EPSV) * g[c];
  const float sh   = be[c] - mean * rs;
  out[r] = fmaxf(0.0f, (float)y5[r] * rs + sh);
}

// ---------------------------------------------------------------------------
extern "C" void kernel_launch(void* const* d_in, const int* in_sizes, int n_in,
                              void* d_out, int out_size, void* d_ws, size_t ws_size,
                              hipStream_t stream) {
  const float* x   = (const float*)d_in[0];
  const float* w1  = (const float*)d_in[1];
  const float* b1  = (const float*)d_in[2];
  const float* g1  = (const float*)d_in[3];
  const float* be1 = (const float*)d_in[4];
  const float* w2  = (const float*)d_in[5];
  const float* b2  = (const float*)d_in[6];
  const float* g2  = (const float*)d_in[7];
  const float* be2 = (const float*)d_in[8];
  const float* w3  = (const float*)d_in[9];
  const float* b3  = (const float*)d_in[10];
  const float* g3  = (const float*)d_in[11];
  const float* be3 = (const float*)d_in[12];
  const float* w4  = (const float*)d_in[13];
  const float* b4  = (const float*)d_in[14];
  const float* g4  = (const float*)d_in[15];
  const float* be4 = (const float*)d_in[16];
  const float* w5  = (const float*)d_in[17];
  const float* b5  = (const float*)d_in[18];
  const float* g5  = (const float*)d_in[19];
  const float* be5 = (const float*)d_in[20];

  // Workspace layout (needs ~580 MB; activations ping-pong in f16):
  //   [0, 256)        : BN stats (5 layers x {sum[3], sumsq[3], pad})
  //   [256, 2176)     : padded biases (5 x 96 f32)
  //   [4096, ~106KB)  : prepacked WMMA B fragments (5 x PB_STRIDE halves)
  //   [1MB, +302MB)   : activation buffer 0
  //   [.. , +302MB)   : activation buffer 1
  char* ws = (char*)d_ws;
  float* stats   = (float*)(ws + 0);
  float* biasPad = (float*)(ws + 256);
  f16*   packB   = (f16*)(ws + 4096);
  const size_t BUFBYTES = (size_t)RTOT * 96 * 2;       // 301,989,888
  f16* buf0 = (f16*)(ws + (1u << 20));
  f16* buf1 = (f16*)(ws + (1u << 20) + BUFBYTES + 10240);

  hipMemsetAsync(stats, 0, 256, stream);
  prepack_kernel<<<1, 256, 0, stream>>>(w1, b1, w2, b2, w3, b3, w4, b4, w5, b5,
                                        packB, biasPad);

  dim3 grid(RTOT / 128), blk(256);
  // L1: patches(25) -> 50            y1 -> buf0 (width 64)
  gemm_layer1<<<grid, blk, 0, stream>>>(x, buf0, packB + 0 * PB_STRIDE,
                                        biasPad + 0, stats + 0);
  // L2: 50 -> 75                     y2 -> buf1 (width 96)
  gemm_layer<50, 75, 64, 96><<<grid, blk, 0, stream>>>(
      buf0, buf1, packB + 1 * PB_STRIDE, biasPad + 96,
      stats + 0, g1, be1, stats + 8, 1.0f / (524288.0f * 50.0f));
  // L3: 75 -> 75                     y3 -> buf0 (width 96)
  gemm_layer<75, 75, 96, 96><<<grid, blk, 0, stream>>>(
      buf1, buf0, packB + 2 * PB_STRIDE, biasPad + 192,
      stats + 8, g2, be2, stats + 16, 1.0f / (524288.0f * 75.0f));
  // L4: 75 -> 50                     y4 -> buf1 (width 64)
  gemm_layer<75, 50, 96, 64><<<grid, blk, 0, stream>>>(
      buf0, buf1, packB + 3 * PB_STRIDE, biasPad + 288,
      stats + 16, g3, be3, stats + 24, 1.0f / (524288.0f * 75.0f));
  // L5: 50 -> 1                      y5 -> buf0 (width 1)
  gemm_layer<50, 1, 64, 1><<<grid, blk, 0, stream>>>(
      buf1, buf0, packB + 4 * PB_STRIDE, biasPad + 384,
      stats + 24, g4, be4, stats + 32, 1.0f / (524288.0f * 50.0f));
  // Final BN + ReLU -> d_out (f32)
  finalize_kernel<<<dim3(RTOT / 256), blk, 0, stream>>>(
      buf0, stats + 32, g5, be5, (float*)d_out);
}
